// MessagePassingEncoder_8615704396355
// MI455X (gfx1250) — matrix operand order repaired and checked by reading, not compiled, not run on hardware
//
#include <hip/hip_runtime.h>
#include <hip/hip_bf16.h>

#define N_NODES 50000
#define N_EDGES 800000
#define IN_CH   128
#define HID     256

typedef __attribute__((ext_vector_type(16))) __bf16          v16bf;
typedef __attribute__((ext_vector_type(8)))  float           v8f;
typedef __attribute__((ext_vector_type(16))) unsigned short  v16u;
typedef __attribute__((ext_vector_type(8)))  unsigned short  v8u;

// ---------- helpers ----------
__device__ __forceinline__ unsigned short f2bf(float f) {
    unsigned int u = __float_as_uint(f);
    unsigned int r = u + 0x7FFFu + ((u >> 16) & 1u);   // round-to-nearest-even
    return (unsigned short)(r >> 16);
}

// A fragment: 16x32 bf16, row-major source [*, ld]; lane<16 K=0..15, lane>=16 K=16..31
__device__ __forceinline__ v16bf load_a_frag(const unsigned short* __restrict__ base,
                                             int row, int ld, int k0, int lane) {
    int kb = k0 + ((lane & 16) ? 16 : 0);
    const unsigned short* p = base + (size_t)row * ld + kb;
    v8u lo = *(const v8u*)(p);        // K = kb .. kb+7
    v8u hi = *(const v8u*)(p + 8);    // K = kb+8 .. kb+15
    v16u u;
#pragma unroll
    for (int i = 0; i < 8; ++i) { u[i] = lo[i]; u[i + 8] = hi[i]; }
    return __builtin_bit_cast(v16bf, u);
}

// B fragment: 32x16 bf16 from K-major (transposed) weights Wt[dout][din];
// lane<16 K=k0..k0+15, lane>=16 K=k0+16..k0+31, N = lane&15  -> contiguous 16 elements
__device__ __forceinline__ v16bf load_b_frag(const unsigned short* __restrict__ Wt,
                                             int col, int ld, int k0, int lane) {
    int kb = k0 + ((lane & 16) ? 16 : 0);
    v16u u = *(const v16u*)(Wt + (size_t)col * ld + kb);
    return __builtin_bit_cast(v16bf, u);
}

// ---------- setup kernels ----------
__global__ void zero_f32(float* __restrict__ p, int n) {
    int i = blockIdx.x * blockDim.x + threadIdx.x;
    if (i < n) p[i] = 0.0f;
}

__global__ void count_edges(const long long* __restrict__ ei, float* __restrict__ cnt, int nE) {
    int e = blockIdx.x * blockDim.x + threadIdx.x;
    if (e < nE) {
        long long d = ei[nE + e];
        atomicAdd(&cnt[d], 1.0f);
    }
}

// Wt[n][k] = bf16(W[k][n]); one thread per element of W (din x dout)
__global__ void transpose_w(const float* __restrict__ W, unsigned short* __restrict__ Wt,
                            int din, int dout) {
    int i = blockIdx.x * blockDim.x + threadIdx.x;
    if (i < din * dout) {
        int k = i / dout;
        int n = i - k * dout;
        Wt[(size_t)n * din + k] = f2bf(W[i]);
    }
}

__global__ void f32_to_bf16(const float* __restrict__ src, unsigned short* __restrict__ dst, int n) {
    int i = blockIdx.x * blockDim.x + threadIdx.x;
    if (i < n) dst[i] = f2bf(src[i]);
}

// ---------- aggregation ----------
// one wave per edge; lanes stride the channels; coalesced gather + L2-resident f32 atomics
__global__ __launch_bounds__(256) void scatter_add(const float* __restrict__ h,
                                                   const long long* __restrict__ ei,
                                                   float* __restrict__ agg,
                                                   int din, int nE) {
    int gtid = blockIdx.x * blockDim.x + threadIdx.x;
    int wave = gtid >> 5;
    int lane = gtid & 31;
    if (wave >= nE) return;
    long long s = ei[wave];
    long long d = ei[nE + wave];
    const float* src = h + (size_t)s * din;
    float*       dst = agg + (size_t)d * din;
    for (int c = lane; c < din; c += 32)
        atomicAdd(&dst[c], src[c]);
}

// agg_bf16[i] = bf16(agg_f32[i] / max(cnt,1))
__global__ void finalize_agg(const float* __restrict__ aggF, const float* __restrict__ cnt,
                             unsigned short* __restrict__ aggB, int din, int total) {
    int i = blockIdx.x * blockDim.x + threadIdx.x;
    if (i < total) {
        int node = i / din;
        float c = cnt[node];
        aggB[i] = f2bf(aggF[i] / fmaxf(c, 1.0f));
    }
}

// ---------- fused SAGE GEMM:  out = relu?( aggB @ Wl^T' + hB @ Wr^T' + b ) ----------
// one wave per 16x16 output tile; M=50000 (3125 tiles), N=256 (16 tiles); K step 32 via bf16 WMMA
__global__ __launch_bounds__(128) void sage_gemm(const unsigned short* __restrict__ aggB,
                                                 const unsigned short* __restrict__ hB,
                                                 const unsigned short* __restrict__ Wlt,
                                                 const unsigned short* __restrict__ Wrt,
                                                 const float* __restrict__ bias,
                                                 float* __restrict__ out,
                                                 int din, int do_relu) {
    int gtid  = blockIdx.x * blockDim.x + threadIdx.x;
    int wave  = gtid >> 5;
    int lane  = gtid & 31;
    int tileM = wave >> 4;     // 16 N-tiles per M row
    int tileN = wave & 15;
    int rowA  = tileM * 16 + (lane & 15);
    int colB  = tileN * 16 + (lane & 15);

    v8f acc = {};
    for (int k0 = 0; k0 < din; k0 += 32) {
        v16bf aAgg = load_a_frag(aggB, rowA, din, k0, lane);
        v16bf bWl  = load_b_frag(Wlt,  colB, din, k0, lane);
        acc = __builtin_amdgcn_wmma_f32_16x16x32_bf16(false, aAgg, false, bWl,
                                                      (short)0, acc, false, false);
        v16bf aH  = load_a_frag(hB,  rowA, din, k0, lane);
        v16bf bWr = load_b_frag(Wrt, colB, din, k0, lane);
        acc = __builtin_amdgcn_wmma_f32_16x16x32_bf16(false, aH, false, bWr,
                                                      (short)0, acc, false, false);
    }

    float bv    = bias[colB];
    int   mhalf = (lane & 16) ? 8 : 0;
#pragma unroll
    for (int r = 0; r < 8; ++r) {
        float v = acc[r] + bv;
        if (do_relu) v = fmaxf(v, 0.0f);
        out[(size_t)(tileM * 16 + mhalf + r) * HID + colB] = v;
    }
}

// ---------- host launch ----------
static inline int cdiv(int a, int b) { return (a + b - 1) / b; }

extern "C" void kernel_launch(void* const* d_in, const int* in_sizes, int n_in,
                              void* d_out, int out_size, void* d_ws, size_t ws_size,
                              hipStream_t stream) {
    const float*     x  = (const float*)d_in[0];
    const long long* ei = (const long long*)d_in[1];
    const float* Wl[3] = { (const float*)d_in[2], (const float*)d_in[5], (const float*)d_in[8]  };
    const float* bb[3] = { (const float*)d_in[3], (const float*)d_in[6], (const float*)d_in[9]  };
    const float* Wr[3] = { (const float*)d_in[4], (const float*)d_in[7], (const float*)d_in[10] };
    float* outp = (float*)d_out;

    // workspace carve-up (256B aligned)
    char*  ws  = (char*)d_ws;
    size_t off = 0;
    auto take = [&](size_t bytes) -> char* {
        char* p = ws + off;
        off = (off + bytes + 255) & ~(size_t)255;
        return p;
    };
    float*          cnt  = (float*)take((size_t)N_NODES * 4);
    float*          aggF = (float*)take((size_t)N_NODES * HID * 4);
    unsigned short* aggB = (unsigned short*)take((size_t)N_NODES * HID * 2);
    unsigned short* hB   = (unsigned short*)take((size_t)N_NODES * HID * 2);
    float*          Hbuf = (float*)take((size_t)N_NODES * HID * 4);
    unsigned short* Wlt[3], *Wrt[3];
    int dins[3] = { IN_CH, HID, HID };
    for (int i = 0; i < 3; ++i) {
        Wlt[i] = (unsigned short*)take((size_t)dins[i] * HID * 2);
        Wrt[i] = (unsigned short*)take((size_t)dins[i] * HID * 2);
    }

    const int B = 256;

    // ---- one-time per launch: counts + bf16 transposed weights ----
    zero_f32<<<cdiv(N_NODES, B), B, 0, stream>>>(cnt, N_NODES);
    count_edges<<<cdiv(N_EDGES, B), B, 0, stream>>>(ei, cnt, N_EDGES);
    for (int i = 0; i < 3; ++i) {
        int ne = dins[i] * HID;
        transpose_w<<<cdiv(ne, B), B, 0, stream>>>(Wl[i], Wlt[i], dins[i], HID);
        transpose_w<<<cdiv(ne, B), B, 0, stream>>>(Wr[i], Wrt[i], dins[i], HID);
    }

    const float* hcur = x;  // layer-0 input features (f32)
    for (int layer = 0; layer < 3; ++layer) {
        int din    = dins[layer];
        int totalA = N_NODES * din;

        // h -> bf16
        f32_to_bf16<<<cdiv(totalA, B), B, 0, stream>>>(hcur, hB, totalA);

        // mean aggregation
        zero_f32<<<cdiv(totalA, B), B, 0, stream>>>(aggF, totalA);
        scatter_add<<<cdiv(N_EDGES * 32, B), B, 0, stream>>>(hcur, ei, aggF, din, N_EDGES);
        finalize_agg<<<cdiv(totalA, B), B, 0, stream>>>(aggF, cnt, aggB, din, totalA);

        // fused dual-GEMM + bias (+ReLU)
        float* dst     = (layer == 2) ? outp : Hbuf;
        int    do_relu = (layer < 2) ? 1 : 0;
        int    nWaves  = (N_NODES / 16) * (HID / 16);   // 3125 * 16 = 50000
        int    nThr    = nWaves * 32;
        sage_gemm<<<nThr / 128, 128, 0, stream>>>(aggB, hB, Wlt[layer], Wrt[layer],
                                                  bb[layer], dst, din, do_relu);
        hcur = Hbuf;
    }
}
